// GRU_78039555768549
// MI455X (gfx1250) — matrix operand order repaired and checked by reading, not compiled, not run on hardware
//
#include <hip/hip_runtime.h>

// GRU for MI455X (gfx1250, wave32, WMMA + TDM).
// Phase 1: xr/xu/xc = x @ W*_x as tiled f16 WMMA GEMM (bandwidth-bound, ~0.5GB).
// Phase 2: 16 persistent workgroups; 3x(512x32) weight slices resident in LDS
//          (~207KB/WG; CDNA5's 320KB WGP LDS) in WMMA fragment order.
//          h is exchanged between steps as a pre-permuted f16 fragment buffer:
//          WG g's 32-column slice == K-tile kt=g of next step's A operand, so
//          staging degenerates to a 48KB DMA -> Tensor Data Mover load to LDS
//          (s_wait_tensorcnt), no per-step conversion/permutation VALU.
//          Grid-wide atomic-counter barrier between steps.
// Requires ws_size >= 3*B*L*D*4 + 2*48KB + 256 bytes (~384 MB) scratch.

typedef _Float16 f16;
typedef __attribute__((ext_vector_type(16))) _Float16 v16h;
typedef __attribute__((ext_vector_type(8)))  _Float16 v8h;
typedef __attribute__((ext_vector_type(4)))  _Float16 v4h;
typedef __attribute__((ext_vector_type(8)))  float    v8f;
typedef __attribute__((ext_vector_type(4)))  unsigned int v4u;
typedef __attribute__((ext_vector_type(8)))  int      v8i;
typedef __attribute__((ext_vector_type(4)))  int      v4i;

#define B_SZ 32
#define L_SZ 2048
#define D_SZ 512
#define K_SZ 512
#define NROW (B_SZ * L_SZ)          // 65536
#define NGRP 16                     // persistent workgroups in phase 2

// Fragment tile storage: [32 lanes][FRAG_LD f16], first 16 elements used.
// FRAG_LD=24 -> per-lane stride 48B -> 16B loads start at banks 12*lane mod 64:
// each half-wave covers all 64 banks exactly once (conflict-free).
#define FRAG_LD 24
#define TILE_F16 (32 * FRAG_LD)     // 768 f16 / 1536 B per fragment tile

#define SX_LD 33                    // phase-2 x-tile row stride (f32)
#define SW_TILES (3 * 16 * 2)       // weights: [3][16 kt][2 nt]
#define SH_TILES (16 * 2)           // h:       [16 kt][2 mt]
#define HFRAG_F16 (SH_TILES * TILE_F16)   // 24576 f16 = 48 KB per h snapshot
#define SMEM_BYTES ((SW_TILES + SH_TILES + 2) * TILE_F16 * 2 + 3 * B_SZ * SX_LD * 4)

#ifndef __has_builtin
#define __has_builtin(x) 0
#endif
#if __has_builtin(__builtin_amdgcn_tensor_load_to_lds) && \
    __has_builtin(__builtin_amdgcn_s_wait_tensorcnt)
#define USE_TDM 1
#else
#define USE_TDM 0
#endif

// ---------------- WMMA fragment layout (ISA 7.12.2, wave32) ----------------
// A 16x32 f16 (MxK): lane = m + 16*((k>>3)&1); elem = 8*(k>>4) + (k&7)
// B 32x16 f16 (KxN): lane = n + 16*((k>>3)&1); elem = 8*(k>>4) + (k&7)
__device__ __forceinline__ int a_lane(int m, int k) { return (m & 15) + (((k >> 3) & 1) << 4); }
__device__ __forceinline__ int a_elem(int k)        { return (((k >> 4) & 1) << 3) + (k & 7); }
__device__ __forceinline__ int b_lane(int k, int n) { return (n & 15) + (((k >> 3) & 1) << 4); }
__device__ __forceinline__ int b_elem(int k)        { return (((k >> 4) & 1) << 3) + (k & 7); }

// Load one lane's 16 contiguous f16 fragment elements (2x ds_load_b128).
__device__ __forceinline__ v16h load_frag(const f16* __restrict__ tile, int lane) {
  const f16* q = tile + lane * FRAG_LD;
  const v8h lo = *(const v8h*)q;
  const v8h hi = *(const v8h*)(q + 8);
  return __builtin_shufflevector(lo, hi, 0, 1, 2, 3, 4, 5, 6, 7,
                                 8, 9, 10, 11, 12, 13, 14, 15);
}

// ---------------- Phase 1: input projections (batched GEMM) ----------------
__global__ __launch_bounds__(256) void proj_gemm(
    const float* __restrict__ X,
    const float* __restrict__ Wr, const float* __restrict__ Wu, const float* __restrict__ Wc,
    float* __restrict__ xr, float* __restrict__ xu, float* __restrict__ xc) {
  __shared__ f16 sA[4 * TILE_F16];        // 4 m-tiles of the 64x32 X slab
  __shared__ f16 sB[12 * TILE_F16];       // 3 weights x 4 n-tiles of 32x64 W slab

  const int row0 = blockIdx.x * 64;
  const int col0 = blockIdx.y * 64;
  const int tid  = threadIdx.x;
  const int wave = tid >> 5, lane = tid & 31;
  const int mi = wave & 3, ni = wave >> 2;

  const float* W[3] = {Wr, Wu, Wc};
  v8f acc[3][2] = {};

  for (int kk = 0; kk < K_SZ; kk += 32) {
    for (int i = tid; i < 64 * 8; i += 256) {
      const int r = i >> 3, k = (i & 7) * 4;
      const float4 v = *(const float4*)&X[(size_t)(row0 + r) * K_SZ + kk + k];
      const int idx = ((r >> 4) * 32 + a_lane(r, k)) * FRAG_LD + a_elem(k);
      const v4h pk = {(f16)v.x, (f16)v.y, (f16)v.z, (f16)v.w};
      *(v4h*)&sA[idx] = pk;
    }
#pragma unroll
    for (int w = 0; w < 3; ++w)
      for (int i = tid; i < 32 * 64; i += 256) {
        const int r = i >> 6, c = i & 63;
        const int idx = ((w * 4 + (c >> 4)) * 32 + b_lane(r, c)) * FRAG_LD + b_elem(r);
        sB[idx] = (f16)W[w][(size_t)(kk + r) * D_SZ + col0 + c];
      }
    __syncthreads();

    const v16h a = load_frag(&sA[mi * TILE_F16], lane);
#pragma unroll
    for (int w = 0; w < 3; ++w)
#pragma unroll
      for (int nt = 0; nt < 2; ++nt) {
        const v16h b = load_frag(&sB[(w * 4 + ni * 2 + nt) * TILE_F16], lane);
        acc[w][nt] = __builtin_amdgcn_wmma_f32_16x16x32_f16(
            false, a, false, b, (short)0, acc[w][nt], false, false);
      }
    __syncthreads();
  }

  float* O[3] = {xr, xu, xc};
#pragma unroll
  for (int w = 0; w < 3; ++w)
#pragma unroll
    for (int nt = 0; nt < 2; ++nt)
#pragma unroll
      for (int v = 0; v < 8; ++v) {
        const int m = mi * 16 + v + ((lane >> 4) << 3);
        const int n = ni * 32 + nt * 16 + (lane & 15);
        O[w][(size_t)(row0 + m) * D_SZ + col0 + n] = acc[w][nt][v];
      }
}

// ---------------- Phase 2: persistent recurrent scan ----------------

__global__ void init_counter(unsigned int* c) { *c = 0u; }

// Issue a 1-D TDM copy of n8*8 bytes from global `src` to LDS `dst`.
// D# per ISA ch.8: group0 {count=1, lds_addr, global_addr, type=2},
// group1 {data_size=8B, tensor_dim0=tile_dim0=n8, dim1=tile1=1, stride0=n8}.
// This toolchain's builtin is the 6-arg form:
//   (uint32x4 g0, int32x8 g1, int32x4, int32x4, int32x8, i32 cpol)
__device__ __forceinline__ void tdm_load_to_lds(const void* src, void* dst, unsigned n8) {
#if USE_TDM
  const unsigned long long ga = (unsigned long long)(uintptr_t)src;
  const unsigned lds = (unsigned)(uintptr_t)dst;   // flat LDS addr[31:0] == LDS offset
  const v4u g0 = {1u, lds, (unsigned)ga,
                  (unsigned)((ga >> 32) & 0x1FFFFFFu) | (2u << 30)};
  const v8i g1 = {0x00030000,                 // data_size=3 (8B), mask=0
                  (int)(n8 << 16),            // tensor_dim0[15:0] in [31:16]
                  (int)((n8 >> 16) | (1u << 16)), // dim0[31:16] | tensor_dim1=1
                  (int)(n8 << 16),            // dim1[31:16]=0 | tile_dim0
                  1,                          // tile_dim1=1, tile_dim2=0
                  (int)n8, 0, 0};             // tensor_dim0_stride
  const v4i gz4 = {0, 0, 0, 0};
  const v8i gz8 = {0, 0, 0, 0, 0, 0, 0, 0};
  __builtin_amdgcn_tensor_load_to_lds(g0, g1, gz4, gz4, gz8, 0);
#else
  const v8h* s = (const v8h*)src;
  v8h* d = (v8h*)dst;
  for (unsigned i = threadIdx.x; i < n8 / 2; i += blockDim.x) d[i] = s[i];
#endif
}

// NGRP workgroups, 128 threads (4 wave32). WG g owns output columns [g*32, g*32+32).
__global__ __launch_bounds__(128) void gru_scan(
    const float* __restrict__ Wr, const float* __restrict__ Wu, const float* __restrict__ Wc,
    const float* __restrict__ xr, const float* __restrict__ xu, const float* __restrict__ xc,
    float* out, f16* hfrag, unsigned int* counter) {
  extern __shared__ __align__(16) char smem_raw[];
  f16*   sW    = (f16*)smem_raw;                   // [3][16 kt][2 nt] fragment tiles
  f16*   sH    = sW + SW_TILES * TILE_F16;         // [16 kt][2 mt] fragment tiles
  f16*   sHout = sH + SH_TILES * TILE_F16;         // [2 mt] tiles: this WG's new h slice
  float* sX    = (float*)(sHout + 2 * TILE_F16);   // [3][32][SX_LD]

  const int g    = blockIdx.x;
  const int col0 = g * 32;
  const int tid  = threadIdx.x;
  const int wave = tid >> 5, lane = tid & 31;
  const int mt = wave & 1, nt = wave >> 1;

  // Preload h-part weight slices (rows [512,1024), cols [col0,col0+32)) in frag order.
  const float* W[3] = {Wr, Wu, Wc};
#pragma unroll
  for (int w = 0; w < 3; ++w) {
    const float* Wh = W[w] + (size_t)K_SZ * D_SZ;
    for (int i = tid; i < K_SZ * 32; i += 128) {
      const int k = i >> 5, n = i & 31;
      const int idx = (((w * 16 + (k >> 5)) * 2 + (n >> 4)) * 32
                       + b_lane(k & 31, n)) * FRAG_LD + b_elem(k);
      sW[idx] = (f16)Wh[(size_t)k * D_SZ + col0 + n];
    }
  }
  const float* P[3] = {xr, xu, xc};

  for (int t = 0; t < L_SZ; ++t) {
    f16* hcur = hfrag + (size_t)(t & 1) * HFRAG_F16;
    // 1) Start h(t-1) fragment DMA into sH (overlaps with x staging below).
    if (t == 0) {
      const v4h z = {(f16)0.f, (f16)0.f, (f16)0.f, (f16)0.f};
      for (int i = tid; i < SH_TILES * TILE_F16 / 4; i += 128)
        *(v4h*)&sH[i * 4] = z;
    } else {
      const f16* hprev = hfrag + (size_t)((t + 1) & 1) * HFRAG_F16;
#if USE_TDM
      if (wave == 0) tdm_load_to_lds(hprev, sH, HFRAG_F16 / 4);
#else
      tdm_load_to_lds(hprev, sH, HFRAG_F16 / 4);
#endif
    }
    // 2) Stage this WG's xr/xu/xc tiles (32x32 each) + prefetch next step's.
#pragma unroll
    for (int w = 0; w < 3; ++w)
      for (int i = tid; i < B_SZ * 32; i += 128) {
        const int b = i >> 5, j = i & 31;
        sX[w * B_SZ * SX_LD + b * SX_LD + j] = P[w][((size_t)b * L_SZ + t) * D_SZ + col0 + j];
      }
    if (t + 1 < L_SZ && tid < B_SZ) {
      __builtin_prefetch(&P[0][((size_t)tid * L_SZ + t + 1) * D_SZ + col0], 0, 0);
      __builtin_prefetch(&P[1][((size_t)tid * L_SZ + t + 1) * D_SZ + col0], 0, 0);
      __builtin_prefetch(&P[2][((size_t)tid * L_SZ + t + 1) * D_SZ + col0], 0, 0);
    }
#if USE_TDM
    if (wave == 0) __builtin_amdgcn_s_wait_tensorcnt((short)0);
#endif
    __syncthreads();

    // 3) Slice GEMMs: (32x512)@(512x32) for r,u,c; this wave's 16x16 quadrant.
    //    Per K-iter: 8x ds_load_b128 + 3x v_wmma (A frag shared by the 3 gates).
    v8f accR = {}, accU = {}, accC = {};
    for (int kt = 0; kt < 16; ++kt) {
      const v16h a  = load_frag(&sH[(kt * 2 + mt) * TILE_F16], lane);
      const v16h bR = load_frag(&sW[((0 * 16 + kt) * 2 + nt) * TILE_F16], lane);
      const v16h bU = load_frag(&sW[((1 * 16 + kt) * 2 + nt) * TILE_F16], lane);
      const v16h bC = load_frag(&sW[((2 * 16 + kt) * 2 + nt) * TILE_F16], lane);
      accR = __builtin_amdgcn_wmma_f32_16x16x32_f16(false, a, false, bR, (short)0, accR, false, false);
      accU = __builtin_amdgcn_wmma_f32_16x16x32_f16(false, a, false, bU, (short)0, accU, false, false);
      accC = __builtin_amdgcn_wmma_f32_16x16x32_f16(false, a, false, bC, (short)0, accC, false, false);
    }

    // 4) Fused gates; fp32 result -> out; f16 fragment-layout slice -> sHout.
#pragma unroll
    for (int v = 0; v < 8; ++v) {
      const int b = mt * 16 + v + ((lane >> 4) << 3);
      const int j = nt * 16 + (lane & 15);
      const float xrv = sX[0 * B_SZ * SX_LD + b * SX_LD + j];
      const float xuv = sX[1 * B_SZ * SX_LD + b * SX_LD + j];
      const float xcv = sX[2 * B_SZ * SX_LD + b * SX_LD + j];
      const float r = 1.f / (1.f + expf(-(xrv + accR[v])));
      const float u = 1.f / (1.f + expf(-(xuv + accU[v])));
      const float c = tanhf(xcv + r * accC[v]);
      // old h for these (b, col0+j): kt == g, kin == j in fragment coords
      const float hold = (float)sH[((g * 2 + (b >> 4)) * 32
                                    + a_lane(b, j)) * FRAG_LD + a_elem(j)];
      const float hnew = u * hold + (1.f - u) * c;
      out[((size_t)b * L_SZ + t) * D_SZ + col0 + j] = hnew;
      sHout[((b >> 4) * 32 + a_lane(b, j)) * FRAG_LD + a_elem(j)] = (f16)hnew;
    }
    __syncthreads();

    // 5) Publish this WG's two fragment tiles (kt=g) into the ping-pong buffer.
    {
      f16* dst = hcur + (size_t)(g * 2) * TILE_F16;
      for (int i = tid; i < 2 * TILE_F16 / 8; i += 128)
        *(v8h*)&dst[i * 8] = *(const v8h*)&sHout[i * 8];
    }

    // 6) Grid-wide step barrier: release stores, arrive, spin, acquire.
    __threadfence();
    __syncthreads();
    if (tid == 0) {
      atomicAdd(counter, 1u);
      const unsigned target = (unsigned)NGRP * (unsigned)(t + 1);
      while (__hip_atomic_load(counter, __ATOMIC_RELAXED, __HIP_MEMORY_SCOPE_AGENT) < target) {
        __builtin_amdgcn_s_sleep(8);
      }
      __threadfence();
    }
    __syncthreads();
  }
}

// ---------------- Host launcher ----------------
extern "C" void kernel_launch(void* const* d_in, const int* in_sizes, int n_in,
                              void* d_out, int out_size, void* d_ws, size_t ws_size,
                              hipStream_t stream) {
  const float* x  = (const float*)d_in[0];
  const float* Wr = (const float*)d_in[1];
  const float* Wu = (const float*)d_in[2];
  const float* Wc = (const float*)d_in[3];
  float* out = (float*)d_out;

  const size_t NB = (size_t)NROW * D_SZ;           // elements per projection buffer
  unsigned int* counter = (unsigned int*)d_ws;
  float* base = (float*)((char*)d_ws + 256);
  float* xr = base;
  float* xu = base + NB;
  float* xc = base + 2 * NB;
  f16* hfrag = (f16*)(base + 3 * NB);              // 2 x 48KB ping-pong h fragments

  // Phase 1: input projections.
  dim3 g1(NROW / 64, D_SZ / 64);                   // 1024 x 8 blocks
  proj_gemm<<<g1, 256, 0, stream>>>(x, Wr, Wu, Wc, xr, xu, xc);

  // Reset grid-barrier counter every launch (deterministic, replay-safe).
  init_counter<<<1, 1, 0, stream>>>(counter);

  // Phase 2: persistent recurrent scan, ~207 KB LDS per WG (CDNA5: 320KB/WG).
  (void)hipFuncSetAttribute((const void*)gru_scan,
                            hipFuncAttributeMaxDynamicSharedMemorySize,
                            (int)SMEM_BYTES);
  gru_scan<<<dim3(NGRP), 128, SMEM_BYTES, stream>>>(Wr, Wu, Wc, xr, xu, xc,
                                                    out, hfrag, counter);
}
